// GAT_38208029066070
// MI455X (gfx1250) — matrix operand order repaired and checked by reading, not compiled, not run on hardware
//
#include <hip/hip_runtime.h>
#include <hip/hip_bf16.h>

#define S_ 500
#define D_ 10
#define T_ 40
#define FT_ 512
#define H_ 64
#define G3_ 192   // 3*H

typedef __attribute__((ext_vector_type(16))) __bf16 v16bf;
typedef __attribute__((ext_vector_type(8)))  float  v8f;

union Frag16 { v16bf v; unsigned u[8]; };

__device__ __forceinline__ unsigned pack_bf16(float a, float b) {
    unsigned ua = __float_as_uint(a), ub = __float_as_uint(b);
    ua += 0x7FFFu + ((ua >> 16) & 1u);
    ub += 0x7FFFu + ((ub >> 16) & 1u);
    return (ua >> 16) | (ub & 0xFFFF0000u);
}

__device__ __forceinline__ float sigmoidf_(float x) { return 1.0f / (1.0f + __expf(-x)); }

// ---- shared GRU scan over T steps; gi [T][192] in LDS; Whh/bhh global (per stock) ----
__device__ void gru_scan_lds(const float* gi, const float* __restrict__ Whh,
                             const float* __restrict__ bhh,
                             float* h, float* gh, float* hs, int T, int tid) {
    if (tid < H_) h[tid] = 0.0f;
    __syncthreads();
    for (int t = 0; t < T; ++t) {
        if (tid < G3_) {
            float acc = bhh[tid];
            #pragma unroll 8
            for (int j = 0; j < H_; ++j) acc += h[j] * Whh[j * G3_ + tid];
            gh[tid] = acc;
        }
        __syncthreads();
        if (tid < H_) {
            const float* g = gi + t * G3_;
            float r  = sigmoidf_(g[tid] + gh[tid]);
            float z  = sigmoidf_(g[H_ + tid] + gh[H_ + tid]);
            float nn = tanhf(g[2 * H_ + tid] + r * gh[2 * H_ + tid]);
            float hn = (1.0f - z) * nn + z * h[tid];
            h[tid] = hn;
            hs[t * H_ + tid] = hn;
        }
        __syncthreads();
    }
}

// ---- scalar additive attention pooling (small T paths): hs [T][64] LDS -> out[64] global ----
__device__ void attn_pool_lds(const float* hs, const float* __restrict__ Wa,
                              const float* __restrict__ va,
                              float* scores, float* __restrict__ out, int T, int tid) {
    if (tid < T) {
        float sc = 0.0f;
        for (int j = 0; j < H_; ++j) {
            float u = 0.0f;
            #pragma unroll 8
            for (int k = 0; k < H_; ++k) u += hs[tid * H_ + k] * Wa[k * H_ + j];
            sc += tanhf(u) * va[j];
        }
        scores[tid] = sc;
    }
    __syncthreads();
    if (tid == 0) {
        float m = -1e30f;
        for (int t = 0; t < T; ++t) m = fmaxf(m, scores[t]);
        float s = 0.0f;
        for (int t = 0; t < T; ++t) { float e = __expf(scores[t] - m); scores[t] = e; s += e; }
        float inv = 1.0f / s;
        for (int t = 0; t < T; ++t) scores[t] *= inv;
    }
    __syncthreads();
    if (tid < H_) {
        float acc = 0.0f;
        for (int t = 0; t < T; ++t) acc += scores[t] * hs[t * H_ + tid];
        out[tid] = acc;
    }
    __syncthreads();
}

// ---- stage one K=32 chunk of A (48x32) and B (32x192) as bf16 into LDS ----
__device__ __forceinline__ void stage_chunk(const float2* __restrict__ X2,
                                            const float2* __restrict__ Wg2,
                                            unsigned* Ab, unsigned* Bb,
                                            int k0, int tid) {
    for (int e = tid; e < 48 * 16; e += 192) {
        int row = e >> 4, dw = e & 15;
        float2 v = make_float2(0.0f, 0.0f);
        if (row < T_) v = X2[row * (FT_ / 2) + (k0 >> 1) + dw];
        Ab[e] = pack_bf16(v.x, v.y);
    }
    for (int e = tid; e < 16 * 192; e += 192) {
        int rp = e / 192, c = e % 192;
        const float* Wg = (const float*)Wg2;
        Bb[e] = pack_bf16(Wg[(size_t)(k0 + 2 * rp) * 192 + c], Wg[(size_t)(k0 + 2 * rp + 1) * 192 + c]);
    }
}

// ============ Kernel 1: fused text encoder, one block (192 thr / 6 waves) per (s,d) ============
// gi = X(40x512) @ Wih(512x192) via bf16 WMMA (double-buffered LDS staging + prefetch),
// then 40-step GRU scan (VALU), then attention pooling with WMMA (48x64x64).
__global__ __launch_bounds__(192) void text_encoder_kernel(
    const float* __restrict__ text,   // (S,D,40,512)
    const float* __restrict__ Wih,    // (S,512,192)
    const float* __restrict__ Whh,    // (S,64,192)
    const float* __restrict__ bih,    // (S,192)
    const float* __restrict__ bhh,    // (S,192)
    const float* __restrict__ Wa,     // (S,64,64)
    const float* __restrict__ va,     // (S,64)
    float* __restrict__ news)         // (S,D,64)
{
    __shared__ unsigned Abuf[2 * 48 * 16];   // ping-pong 48x32 bf16 (reused for hs-bf16 in pool)
    __shared__ unsigned Bbuf[2 * 32 * 96];   // ping-pong 32x192 bf16 (reused for Wa-bf16 in pool)
    __shared__ float    gi[48 * G3_];        // reused as U (48x64) in pool
    __shared__ float    gh[G3_];
    __shared__ float    hcur[H_];
    __shared__ float    hs[T_ * H_];
    __shared__ float    scores[64];
    __shared__ float    bih_l[G3_];

    const int blk = blockIdx.x;
    const int s = blk / D_, d = blk % D_;
    const int tid = threadIdx.x;
    const int lane = tid & 31, wave = tid >> 5;
    const int hi = lane >> 4;

    const float*  X   = text + (size_t)(s * D_ + d) * T_ * FT_;
    const float2* X2  = (const float2*)X;
    const float*  Wg  = Wih + (size_t)s * FT_ * G3_;
    const float2* Wg2 = (const float2*)Wg;
    const float*  vas = va + s * H_;

    if (tid < G3_) bih_l[tid] = bih[s * G3_ + tid];

    v8f acc[6];
    #pragma unroll
    for (int i = 0; i < 6; ++i)
        #pragma unroll
        for (int v = 0; v < 8; ++v) acc[i][v] = 0.0f;

    // prologue: stage chunk 0
    stage_chunk(X2, Wg2, Abuf, Bbuf, 0, tid);
    __syncthreads();

    for (int kc = 0; kc < 16; ++kc) {
        const int cur = kc & 1, nxt = cur ^ 1;
        if (kc + 1 < 16) {
            const int k1 = (kc + 1) * 32;
            stage_chunk(X2, Wg2, Abuf + nxt * (48 * 16), Bbuf + nxt * (32 * 96), k1, tid);
            // prefetch chunk kc+2 toward L2 while kc computes and kc+1 stages
            if (kc + 2 < 16) {
                const int k2 = (kc + 2) * 32;
                __builtin_prefetch(Wg + (size_t)k2 * G3_ + tid * 32, 0, 1);
                if (tid < T_) __builtin_prefetch(X + tid * FT_ + k2, 0, 1);
            }
        }
        const unsigned* Ab = Abuf + cur * (48 * 16);
        const unsigned* Bb = Bbuf + cur * (32 * 96);
        // 6 tiles per wave, fully uniform (no exec masking around WMMA)
        #pragma unroll
        for (int i = 0; i < 6; ++i) {
            const int t = wave * 6 + i;
            const int m = t / 12, n = t % 12;
            Frag16 a, b;
            const int arow = m * 16 + (lane & 15);
            #pragma unroll
            for (int v = 0; v < 8; ++v) {
                int base = (v < 4) ? v : (v + 4);
                a.u[v] = Ab[arow * 16 + base + hi * 4];
            }
            #pragma unroll
            for (int v = 0; v < 8; ++v)
                b.u[v] = Bb[(((v < 4) ? v : (v + 4)) + hi * 4) * 192 + n * 16 + (lane & 15)];
            acc[i] = __builtin_amdgcn_wmma_f32_16x16x32_bf16(
                false, a.v, false, b.v, (short)0, acc[i], false, false);
        }
        __syncthreads();
    }

    // write gi (+bias) to LDS per the C/D layout
    #pragma unroll
    for (int i = 0; i < 6; ++i) {
        const int t = wave * 6 + i;
        const int m = t / 12, n = t % 12;
        const int nl = lane & 15, mh = hi * 8;
        #pragma unroll
        for (int v = 0; v < 8; ++v) {
            int row = m * 16 + mh + v;
            int col = n * 16 + nl;
            gi[row * G3_ + col] = acc[i][v] + bih_l[col];
        }
    }
    __syncthreads();

    gru_scan_lds(gi, Whh + (size_t)s * H_ * G3_, bhh + s * G3_, hcur, gh, hs, T_, tid);

    // ---- WMMA attention pooling: U = hs(48x64,pad) @ Wa(64x64) ----
    float* U = gi;  // reuse
    // stage hs -> bf16 (48 rows x 32 dwords) into Abuf
    for (int e = tid; e < 48 * 32; e += 192) {
        int row = e >> 5, col = (e & 31) * 2;
        float a0 = 0.0f, a1 = 0.0f;
        if (row < T_) { a0 = hs[row * H_ + col]; a1 = hs[row * H_ + col + 1]; }
        Abuf[e] = pack_bf16(a0, a1);
    }
    // stage Wa -> bf16 (64 rows x 32 dwords) into Bbuf
    {
        const float* Was = Wa + (size_t)s * H_ * H_;
        for (int e = tid; e < 32 * 64; e += 192) {
            int rp = e >> 6, c = e & 63;
            Bbuf[e] = pack_bf16(Was[(2 * rp) * H_ + c], Was[(2 * rp + 1) * H_ + c]);
        }
    }
    if (tid < 48) scores[tid] = 0.0f;
    __syncthreads();
    // 12 tiles (3x4) over 6 waves -> 2 tiles/wave, K = 2 chunks of 32
    #pragma unroll
    for (int i = 0; i < 2; ++i) {
        const int t = wave * 2 + i;
        const int m = t >> 2, n = t & 3;
        v8f pacc;
        #pragma unroll
        for (int v = 0; v < 8; ++v) pacc[v] = 0.0f;
        #pragma unroll
        for (int kc2 = 0; kc2 < 2; ++kc2) {
            Frag16 a, b;
            const int arow = m * 16 + (lane & 15);
            #pragma unroll
            for (int v = 0; v < 8; ++v) {
                int base = (v < 4) ? v : (v + 4);
                a.u[v] = Abuf[arow * 32 + kc2 * 16 + base + hi * 4];
            }
            #pragma unroll
            for (int v = 0; v < 8; ++v)
                b.u[v] = Bbuf[(kc2 * 16 + ((v < 4) ? v : (v + 4)) + hi * 4) * 64 + n * 16 + (lane & 15)];
            pacc = __builtin_amdgcn_wmma_f32_16x16x32_bf16(
                false, a.v, false, b.v, (short)0, pacc, false, false);
        }
        const int nl = lane & 15, mh = hi * 8;
        #pragma unroll
        for (int v = 0; v < 8; ++v)
            U[(m * 16 + mh + v) * H_ + n * 16 + nl] = pacc[v];
    }
    __syncthreads();
    // scores[t] = sum_j tanh(U[t][j]) * va[j]  (LDS float atomics across 192 threads)
    for (int e = tid; e < T_ * H_; e += 192) {
        int t = e >> 6, j = e & 63;
        atomicAdd(&scores[t], tanhf(U[e]) * vas[j]);
    }
    __syncthreads();
    if (tid == 0) {
        float m = -1e30f;
        for (int t = 0; t < T_; ++t) m = fmaxf(m, scores[t]);
        float ssum = 0.0f;
        for (int t = 0; t < T_; ++t) { float e = __expf(scores[t] - m); scores[t] = e; ssum += e; }
        float inv = 1.0f / ssum;
        for (int t = 0; t < T_; ++t) scores[t] *= inv;
    }
    __syncthreads();
    if (tid < H_) {
        float accp = 0.0f;
        for (int t = 0; t < T_; ++t) accp += scores[t] * hs[t * H_ + tid];
        news[(size_t)(s * D_ + d) * H_ + tid] = accp;
    }
}

// ============ Kernel 2: price encoder, one block per stock ============
__global__ __launch_bounds__(192) void price_encoder_kernel(
    const float* __restrict__ price, const float* __restrict__ Wih,
    const float* __restrict__ Whh, const float* __restrict__ bih,
    const float* __restrict__ bhh, const float* __restrict__ Wa,
    const float* __restrict__ va, float* __restrict__ out)
{
    __shared__ float gi[D_ * G3_];
    __shared__ float gh[G3_], hcur[H_], hs[D_ * H_], scores[16];
    const int s = blockIdx.x, tid = threadIdx.x;
    if (tid < G3_) {
        float bi = bih[s * G3_ + tid];
        for (int t = 0; t < D_; ++t) {
            float acc = bi;
            #pragma unroll
            for (int f = 0; f < 3; ++f)
                acc += price[(s * D_ + t) * 3 + f] * Wih[(size_t)s * 3 * G3_ + f * G3_ + tid];
            gi[t * G3_ + tid] = acc;
        }
    }
    __syncthreads();
    gru_scan_lds(gi, Whh + (size_t)s * H_ * G3_, bhh + s * G3_, hcur, gh, hs, D_, tid);
    attn_pool_lds(hs, Wa + (size_t)s * H_ * H_, va + s * H_, scores, out + s * H_, D_, tid);
}

// ============ Kernel 3: day-sequence encoder over news, one block per stock ============
__global__ __launch_bounds__(192) void seq_encoder_kernel(
    const float* __restrict__ news, const float* __restrict__ Wih,
    const float* __restrict__ Whh, const float* __restrict__ bih,
    const float* __restrict__ bhh, const float* __restrict__ Wa,
    const float* __restrict__ va, float* __restrict__ out)
{
    __shared__ float gi[D_ * G3_];
    __shared__ float gh[G3_], hcur[H_], hs[D_ * H_], scores[16];
    const int s = blockIdx.x, tid = threadIdx.x;
    if (tid < G3_) {
        float bi = bih[s * G3_ + tid];
        for (int t = 0; t < D_; ++t) {
            float acc = bi;
            #pragma unroll 8
            for (int j = 0; j < H_; ++j)
                acc += news[(size_t)(s * D_ + t) * H_ + j] * Wih[(size_t)s * H_ * G3_ + j * G3_ + tid];
            gi[t * G3_ + tid] = acc;
        }
    }
    __syncthreads();
    gru_scan_lds(gi, Whh + (size_t)s * H_ * G3_, bhh + s * G3_, hcur, gh, hs, D_, tid);
    attn_pool_lds(hs, Wa + (size_t)s * H_ * H_, va + s * H_, scores, out + s * H_, D_, tid);
}

// ============ Kernel 4: bilinear fusion (streams 524MB Wb, coalesced float2) + blend head ============
__global__ __launch_bounds__(256) void bilinear_kernel(
    const float* __restrict__ tv, const float* __restrict__ xp,
    const float* __restrict__ Wb, const float* __restrict__ bb,
    const float* __restrict__ Wbl, const float* __restrict__ bbl,
    float* __restrict__ combined, float* __restrict__ out1)
{
    __shared__ float tvl[H_], xpl[H_], combl[H_];
    const int s = blockIdx.x, tid = threadIdx.x;
    const int lane = tid & 31, wave = tid >> 5;
    if (tid < H_) { tvl[tid] = tv[s * H_ + tid]; xpl[tid] = xp[s * H_ + tid]; }
    __syncthreads();
    const float x0 = xpl[2 * lane], x1 = xpl[2 * lane + 1];
    const float* W = Wb + (size_t)s * H_ * H_ * H_;
    for (int k = wave; k < H_; k += 8) {
        float acc = 0.0f;
        const float* Wk_ = W + (size_t)k * H_ * H_;
        for (int i = 0; i < H_; ++i) {
            float2 w = *(const float2*)(Wk_ + i * H_ + 2 * lane);
            acc += tvl[i] * (w.x * x0 + w.y * x1);
        }
        #pragma unroll
        for (int off = 16; off > 0; off >>= 1) acc += __shfl_down(acc, off, 32);
        if (lane == 0) combl[k] = tanhf(acc + bb[s * H_ + k]);
    }
    __syncthreads();
    if (tid < H_) combined[s * H_ + tid] = combl[tid];
    if (tid < 2) {
        float acc = bbl[s * 2 + tid];
        for (int i = 0; i < H_; ++i) acc += combl[i] * Wbl[(size_t)s * H_ * 2 + i * 2 + tid];
        out1[s * 2 + tid] = tanhf(acc);
    }
}

// ============ Kernel 5: shared q/k projections ============
__global__ __launch_bounds__(128) void qk_kernel(
    const float* __restrict__ comb, const float* __restrict__ Wq, const float* __restrict__ bq,
    const float* __restrict__ Wk, const float* __restrict__ bk,
    float* __restrict__ qv, float* __restrict__ kv)
{
    __shared__ float c[H_];
    const int s = blockIdx.x, tid = threadIdx.x;
    if (tid < H_) c[tid] = comb[s * H_ + tid];
    __syncthreads();
    if (tid < H_) {
        float acc = bq[tid];
        for (int i = 0; i < H_; ++i) acc += c[i] * Wq[i * H_ + tid];
        qv[s * H_ + tid] = acc;
    } else {
        int j = tid - H_;
        float acc = bk[j];
        for (int i = 0; i < H_; ++i) acc += c[i] * Wk[i * H_ + j];
        kv[s * H_ + j] = acc;
    }
}

// ============ Kernel 6: cross-stock MHSA, one block per (s, head) ============
__global__ __launch_bounds__(256) void xattn_kernel(
    const float* __restrict__ qv, const float* __restrict__ kv,
    const float* __restrict__ comb, float* __restrict__ attn_out)
{
    __shared__ float sc[S_];
    __shared__ float red[256];
    const int b = blockIdx.x, s = b >> 2, h = b & 3, tid = threadIdx.x;
    float q[16];
    #pragma unroll
    for (int d2 = 0; d2 < 16; ++d2) q[d2] = qv[s * H_ + h * 16 + d2];
    float lmax = -1e30f;
    for (int t = tid; t < S_; t += 256) {
        float dot = 0.0f;
        #pragma unroll
        for (int d2 = 0; d2 < 16; ++d2) dot += q[d2] * kv[t * H_ + h * 16 + d2];
        dot *= 0.25f;            // 1/sqrt(16)
        sc[t] = dot;
        lmax = fmaxf(lmax, dot);
    }
    red[tid] = lmax; __syncthreads();
    for (int o = 128; o > 0; o >>= 1) { if (tid < o) red[tid] = fmaxf(red[tid], red[tid + o]); __syncthreads(); }
    const float m = red[0];
    __syncthreads();
    float lsum = 0.0f;
    for (int t = tid; t < S_; t += 256) { float e = __expf(sc[t] - m); sc[t] = e; lsum += e; }
    red[tid] = lsum; __syncthreads();
    for (int o = 128; o > 0; o >>= 1) { if (tid < o) red[tid] += red[tid + o]; __syncthreads(); }
    const float inv = 1.0f / red[0];
    __syncthreads();
    // attn_out = alpha @ V  (V = combined head slice)
    {
        int part = tid >> 4, dd = tid & 15;
        float acc = 0.0f;
        int t0 = part * 32, t1 = min(S_, t0 + 32);
        for (int t = t0; t < t1; ++t) acc += sc[t] * comb[t * H_ + h * 16 + dd];
        red[tid] = acc;
    }
    __syncthreads();
    if (tid < 16) {
        float acc = 0.0f;
        #pragma unroll
        for (int p = 0; p < 16; ++p) acc += red[p * 16 + tid];
        attn_out[s * H_ + h * 16 + tid] = acc * inv;
    }
}

// ============ Kernel 7: head + double softmax + CE loss ============
__global__ __launch_bounds__(512) void head_kernel(
    const float* __restrict__ ao, const float* __restrict__ out1,
    const float* __restrict__ Wf, const float* __restrict__ bfv,
    const int* __restrict__ label, float* __restrict__ d_out)
{
    __shared__ float lossbuf[512];
    const int tid = threadIdx.x;
    float l = 0.0f;
    if (tid < S_) {
        const int s = tid;
        float x0 = bfv[0], x1 = bfv[1];
        for (int i = 0; i < H_; ++i) {
            float a = ao[s * H_ + i];
            x0 += a * Wf[i * 2];
            x1 += a * Wf[i * 2 + 1];
        }
        x0 = (x0 > 0.0f) ? x0 : (__expf(x0) - 1.0f);   // elu
        x1 = (x1 > 0.0f) ? x1 : (__expf(x1) - 1.0f);
        x0 += out1[s * 2]; x1 += out1[s * 2 + 1];
        float m = fmaxf(x0, x1);
        float e0 = __expf(x0 - m), e1 = __expf(x1 - m);
        float inv = 1.0f / (e0 + e1);
        float p0 = e0 * inv, p1 = e1 * inv;
        d_out[1 + s * 2]     = p0;
        d_out[1 + s * 2 + 1] = p1;
        // faithful: log_softmax of the already-softmaxed output
        float mm  = fmaxf(p0, p1);
        float lse = mm + __logf(__expf(p0 - mm) + __expf(p1 - mm));
        float lp  = ((label[s] == 0) ? p0 : p1) - lse;
        l = -lp;
    }
    lossbuf[tid] = l; __syncthreads();
    for (int o = 256; o > 0; o >>= 1) { if (tid < o) lossbuf[tid] += lossbuf[tid + o]; __syncthreads(); }
    if (tid == 0) d_out[0] = lossbuf[0] / (float)S_;
}

extern "C" void kernel_launch(void* const* d_in, const int* in_sizes, int n_in,
                              void* d_out, int out_size, void* d_ws, size_t ws_size,
                              hipStream_t stream)
{
    const float* text  = (const float*)d_in[0];
    const float* price = (const float*)d_in[1];
    const int*   label = (const int*)d_in[2];
    // d_in[3]=adj, d_in[4]=train unused by the forward pass
    const float* Wih_p = (const float*)d_in[5];
    const float* Whh_p = (const float*)d_in[6];
    const float* bih_p = (const float*)d_in[7];
    const float* bhh_p = (const float*)d_in[8];
    const float* Wa_p  = (const float*)d_in[9];
    const float* va_p  = (const float*)d_in[10];
    const float* Wih_t = (const float*)d_in[11];
    const float* Whh_t = (const float*)d_in[12];
    const float* bih_t = (const float*)d_in[13];
    const float* bhh_t = (const float*)d_in[14];
    const float* Wa_t  = (const float*)d_in[15];
    const float* va_t  = (const float*)d_in[16];
    const float* Wih_s = (const float*)d_in[17];
    const float* Whh_s = (const float*)d_in[18];
    const float* bih_s = (const float*)d_in[19];
    const float* bhh_s = (const float*)d_in[20];
    const float* Wa_s  = (const float*)d_in[21];
    const float* va_s  = (const float*)d_in[22];
    const float* Wb    = (const float*)d_in[23];
    const float* bb    = (const float*)d_in[24];
    const float* Wbl   = (const float*)d_in[25];
    const float* bbl   = (const float*)d_in[26];
    const float* Wq    = (const float*)d_in[27];
    const float* bq    = (const float*)d_in[28];
    const float* Wk    = (const float*)d_in[29];
    const float* bk    = (const float*)d_in[30];
    const float* Wf    = (const float*)d_in[31];
    const float* bfv   = (const float*)d_in[32];

    float* ws       = (float*)d_ws;
    float* news     = ws;                 // S*D*64 = 320000
    float* x_price  = ws + 320000;        // 32000
    float* text_vec = ws + 352000;        // 32000
    float* combined = ws + 384000;        // 32000
    float* out1     = ws + 416000;        // 1000
    float* qv       = ws + 417000;        // 32000
    float* kv       = ws + 449000;        // 32000
    float* ao       = ws + 481000;        // 32000  (end: 513000 floats ~= 2.05 MB)

    hipLaunchKernelGGL(text_encoder_kernel, dim3(S_ * D_), dim3(192), 0, stream,
                       text, Wih_t, Whh_t, bih_t, bhh_t, Wa_t, va_t, news);
    hipLaunchKernelGGL(price_encoder_kernel, dim3(S_), dim3(192), 0, stream,
                       price, Wih_p, Whh_p, bih_p, bhh_p, Wa_p, va_p, x_price);
    hipLaunchKernelGGL(seq_encoder_kernel, dim3(S_), dim3(192), 0, stream,
                       news, Wih_s, Whh_s, bih_s, bhh_s, Wa_s, va_s, text_vec);
    hipLaunchKernelGGL(bilinear_kernel, dim3(S_), dim3(256), 0, stream,
                       text_vec, x_price, Wb, bb, Wbl, bbl, combined, out1);
    hipLaunchKernelGGL(qk_kernel, dim3(S_), dim3(128), 0, stream,
                       combined, Wq, bq, Wk, bk, qv, kv);
    hipLaunchKernelGGL(xattn_kernel, dim3(S_ * 4), dim3(256), 0, stream,
                       qv, kv, combined, ao);
    hipLaunchKernelGGL(head_kernel, dim3(1), dim3(512), 0, stream,
                       ao, out1, Wf, bfv, label, (float*)d_out);
}
